// CustomNetwork_6897717477418
// MI455X (gfx1250) — compile-verified, matrix operand order checked by reading, not run on hardware
//
#include <hip/hip_runtime.h>

// ---------------------------------------------------------------------------
// GraphNet MetaLayer inference for MI455X (gfx1250, wave32, WMMA 16x16x32 f16)
// - fused edge->node1->scatter kernel (no 157MB intermediates; agg in L2)
// - weights pre-swizzled to WMMA B-fragment order (1 contiguous 32B load/frag)
// - activations staged in LDS in A-fragment order (1 contiguous 32B load/frag)
// ---------------------------------------------------------------------------

#define NODES_C 120
#define B_C     256
#define E_C     1200
#define NE_C    (B_C * E_C)     // 307200 edges
#define NN_C    (B_C * NODES_C) // 30720 nodes
#define FROW    4206            // 5*120 + 6 + 3*1200

typedef __attribute__((ext_vector_type(16))) _Float16 v16h;
typedef __attribute__((ext_vector_type(8)))  float    v8f;

// K held by (lane, element i) of a 16-bit A/B fragment (CDNA5 ISA 7.12.2):
// lanes 0-15: K 0-7 (v0-3), 16-23 (v4-7); lanes 16-31: K 8-15, 24-31.
__device__ __forceinline__ int frag_k(int lane, int i) {
  int vg = i >> 1, j = i & 1;
  int kb = (lane < 16) ? 0 : 8;
  int base = (vg < 4) ? (kb + 2 * vg) : (16 + kb + 2 * (vg - 4));
  return base + j;
}

// Inverse: K offset (0..31) -> (lane half, element index)
__device__ __forceinline__ void inv_frag(int ko, int& half, int& i) {
  half = (ko >> 3) & 1;
  int vg = ((ko & 7) >> 1) + ((ko >> 4) << 2);
  i = vg * 2 + (ko & 1);
}

__device__ __forceinline__ v8f wmma16(v16h a, v16h b, v8f c) {
  return __builtin_amdgcn_wmma_f32_16x16x32_f16(
      false, a, false, b, (short)0, c, false, false);
}

__device__ __forceinline__ v16h ldsA(const _Float16* p) {
  return *(const v16h*)p;
}
__device__ __forceinline__ v16h glbB(const _Float16* p) {
  return *(const v16h*)p;
}

// ------------------------------- prep kernels ------------------------------

__global__ void k_zero(float* __restrict__ p, int n) {
  int i = blockIdx.x * blockDim.x + threadIdx.x;
  if (i < n) p[i] = 0.f;
}

// Swizzle fp32 KxN row-major weight into WMMA B-fragment order (f16, padded):
// dst[((nt*KB + kb)*32 + lane)*16 + i] = W[kb*32+frag_k(lane,i)][nt*16+(lane&15)]
__global__ void k_swz(const float* __restrict__ w, _Float16* __restrict__ d,
                      int K, int N, int KB, int NT) {
  int idx = blockIdx.x * blockDim.x + threadIdx.x;
  int total = NT * KB * 512;
  if (idx >= total) return;
  int i = idx & 15;
  int lane = (idx >> 4) & 31;
  int kb = (idx >> 9) % KB;
  int nt = idx / (KB * 512);
  int k = kb * 32 + frag_k(lane, i);
  int n = nt * 16 + (lane & 15);
  float v = (k < K && n < N) ? w[k * N + n] : 0.f;
  d[idx] = (_Float16)v;
}

__global__ void k_build_x0(const float* __restrict__ F, float* __restrict__ x0) {
  int i = blockIdx.x * blockDim.x + threadIdx.x;
  if (i >= NN_C) return;
  int b = i / NODES_C, j = i % NODES_C;
  const float* f = F + (long)b * FROW;
  x0[i * 5 + 0] = f[NODES_C + j];      // capacities
  x0[i * 5 + 1] = f[j];                // degrees
  x0[i * 5 + 2] = f[2 * NODES_C + j];  // inc
  x0[i * 5 + 3] = f[3 * NODES_C + j];  // out
  x0[i * 5 + 4] = f[4 * NODES_C + j];  // tot
}

__global__ void k_build_edges(const float* __restrict__ F, float* __restrict__ e0,
                              int* __restrict__ row, int* __restrict__ col,
                              float* __restrict__ deg) {
  int i = blockIdx.x * blockDim.x + threadIdx.x;
  if (i >= NE_C) return;
  int b = i / E_C, e = i % E_C;
  const float* f = F + (long)b * FROW + 5 * NODES_C + 6;
  e0[i] = f[e];
  int r = (int)f[E_C + e] + b * NODES_C;
  int c = (int)f[2 * E_C + e] + b * NODES_C;
  row[i] = r;
  col[i] = c;
  atomicAdd(&deg[c], 1.0f);
}

__global__ void k_build_u0(const float* __restrict__ F, float* __restrict__ u0) {
  int i = blockIdx.x * blockDim.x + threadIdx.x;
  if (i >= B_C * 6) return;
  int b = i / 6, k = i % 6;
  u0[i] = F[(long)b * FROW + 5 * NODES_C + k];
}

// ------------------- fused edge MLP + node1 MLP + scatter -------------------
// 256 threads = 8 waves; wave owns a 16-edge tile. Four WMMA GEMMs:
//   H1 = relu([x[row],x[col],ea,u] @ We1 + b)  (K1 = 2DX+DE+DU, N=128)
//   e2 = H1 @ We2 + b                           (K=128, one N tile)
//   H2 = relu([x[row],e2] @ Wn1 + b)            (K2 = DX+DE <= 32)
//   m  = H2 @ Wn2 + b ; atomicAdd into agg[col] (K=128, N=128)
template <int DX, int DE, int DU, bool WRITE_E2>
__launch_bounds__(256)
__global__ void edge_node1_kernel(
    const float* __restrict__ x,
    const float* __restrict__ ea,
    const float* __restrict__ u,
    const int* __restrict__ row, const int* __restrict__ col,
    const _Float16* __restrict__ wfe1, const float* __restrict__ be1,
    const _Float16* __restrict__ wfe2, const float* __restrict__ be2,
    const _Float16* __restrict__ wfn1, const float* __restrict__ bn1,
    const _Float16* __restrict__ wfn2, const float* __restrict__ bn2,
    float* __restrict__ e2_out, float* __restrict__ agg) {
  constexpr int K1 = 2 * DX + DE + DU;
  constexpr int K2 = DX + DE;
  constexpr int KB1 = (K1 + 31) / 32;

  __shared__ __align__(32) _Float16 sEinF[8][KB1][512];  // A-frag layout input
  __shared__ __align__(32) _Float16 sHf[8][4][512];      // A-frag layout hidden
  __shared__ float sE2[8][16 * DE];
  __shared__ int sRow[128], sCol[128];

  const int tid = threadIdx.x;
  const int wave = tid >> 5, lane = tid & 31;
  const int m = lane & 15;
  const int base = blockIdx.x * 128;

  __builtin_prefetch(wfn2, 0, 1);  // global_prefetch_b8: warm L2
  if (tid < 128) { sRow[tid] = row[base + tid]; sCol[tid] = col[base + tid]; }
  {  // zero-fill fragment-layout input (covers K padding)
    int* z = (int*)&sEinF[0][0][0];
    for (int i = tid; i < 8 * KB1 * 256; i += 256) z[i] = 0;
  }
  __syncthreads();

  // cooperative build of edge-input in A-fragment order
  for (int idx = tid; idx < 128 * K1; idx += 256) {
    int e = idx / K1, k = idx - e * K1;
    int r = sRow[e], c = sCol[e];
    float v;
    if (k < DX)               v = x[r * DX + k];
    else if (k < 2 * DX)      v = x[c * DX + (k - DX)];
    else if (k < 2 * DX + DE) v = ea[(base + e) * DE + (k - 2 * DX)];
    else                      v = u[((base + e) / E_C) * DU + (k - 2 * DX - DE)];
    int kb = k >> 5, ko = k & 31, h, ii;
    inv_frag(ko, h, ii);
    sEinF[e >> 4][kb][((e & 15) + 16 * h) * 16 + ii] = (_Float16)v;
  }
  __syncthreads();

  v8f acc[8];
#pragma unroll
  for (int t = 0; t < 8; ++t) acc[t] = {};

  // GEMM 1: edge-input @ We1
#pragma unroll
  for (int kb = 0; kb < KB1; ++kb) {
    v16h a = ldsA(&sEinF[wave][kb][lane * 16]);
#pragma unroll
    for (int nt = 0; nt < 8; ++nt) {
      v16h b = glbB(&wfe1[((nt * KB1 + kb) * 32 + lane) * 16]);
      acc[nt] = wmma16(a, b, acc[nt]);
    }
  }
  // bias + relu -> LDS in A-fragment order
#pragma unroll
  for (int nt = 0; nt < 8; ++nt) {
    float bias = be1[nt * 16 + m];
    int n = nt * 16 + m, kb = n >> 5, ko = n & 31, h, ii;
    inv_frag(ko, h, ii);
#pragma unroll
    for (int rr = 0; rr < 8; ++rr) {
      int mrow = (lane < 16) ? rr : (8 + rr);
      float v = acc[nt][rr] + bias;
      sHf[wave][kb][(mrow + 16 * h) * 16 + ii] = (_Float16)(v > 0.f ? v : 0.f);
    }
  }

  // GEMM 2: e2 = H1 @ We2 (one N tile, cols < DE valid)
  {
    v8f a2 = {};
#pragma unroll
    for (int kb = 0; kb < 4; ++kb) {
      v16h a = ldsA(&sHf[wave][kb][lane * 16]);
      v16h b = glbB(&wfe2[(kb * 32 + lane) * 16]);
      a2 = wmma16(a, b, a2);
    }
    float bias2 = (m < DE) ? be2[m] : 0.f;
#pragma unroll
    for (int rr = 0; rr < 8; ++rr) {
      int mrow = (lane < 16) ? rr : (8 + rr);
      if (m < DE) {
        float v = a2[rr] + bias2;
        sE2[wave][mrow * DE + m] = v;
        if (WRITE_E2) e2_out[(long)(base + wave * 16 + mrow) * DE + m] = v;
      }
    }
  }

  // GEMM 3: H2 = relu([x[row], e2] @ Wn1 + b)  (K2 <= 32: one k-step)
  {
    const int r = sRow[wave * 16 + m];
    v16h a3;
#pragma unroll
    for (int i = 0; i < 16; ++i) {
      int k = frag_k(lane, i);
      float v = 0.f;
      if (k < DX)      v = x[r * DX + k];
      else if (k < K2) v = sE2[wave][m * DE + (k - DX)];
      a3[i] = (_Float16)v;
    }
#pragma unroll
    for (int t = 0; t < 8; ++t) acc[t] = {};
#pragma unroll
    for (int nt = 0; nt < 8; ++nt) {
      v16h b = glbB(&wfn1[(nt * 32 + lane) * 16]);
      acc[nt] = wmma16(a3, b, acc[nt]);
    }
  }
#pragma unroll
  for (int nt = 0; nt < 8; ++nt) {
    float bias = bn1[nt * 16 + m];
    int n = nt * 16 + m, kb = n >> 5, ko = n & 31, h, ii;
    inv_frag(ko, h, ii);
#pragma unroll
    for (int rr = 0; rr < 8; ++rr) {
      int mrow = (lane < 16) ? rr : (8 + rr);
      float v = acc[nt][rr] + bias;
      sHf[wave][kb][(mrow + 16 * h) * 16 + ii] = (_Float16)(v > 0.f ? v : 0.f);
    }
  }

  // GEMM 4: message = H2 @ Wn1_2 + b, scatter-add into agg[col]
#pragma unroll
  for (int t = 0; t < 8; ++t) acc[t] = {};
#pragma unroll
  for (int kb = 0; kb < 4; ++kb) {
    v16h a = ldsA(&sHf[wave][kb][lane * 16]);
#pragma unroll
    for (int nt = 0; nt < 8; ++nt) {
      v16h b = glbB(&wfn2[((nt * 4 + kb) * 32 + lane) * 16]);
      acc[nt] = wmma16(a, b, acc[nt]);
    }
  }
#pragma unroll
  for (int nt = 0; nt < 8; ++nt) {
    float bias = bn2[nt * 16 + m];
#pragma unroll
    for (int rr = 0; rr < 8; ++rr) {
      int mrow = (lane < 16) ? rr : (8 + rr);
      int cdst = sCol[wave * 16 + mrow];
      atomicAdd(&agg[cdst * 128 + nt * 16 + m], acc[nt][rr] + bias);
    }
  }
}

// ------------------------------- node2 MLP ---------------------------------
// 128 threads = 4 waves; wave owns 16 nodes. Hn = relu(in @ W1 + b) with
// N=256 (16 tiles, single reused accumulator), X2 = Hn @ W2 + b (one tile).
template <int DX, int DOUT, bool WRITE_X, bool WRITE_NM, bool WRITE_OUT>
__launch_bounds__(128)
__global__ void node2_kernel(
    const float* __restrict__ x,
    const float* __restrict__ agg, const float* __restrict__ deg,
    const _Float16* __restrict__ wf1, const float* __restrict__ b1,
    const _Float16* __restrict__ wf2, const float* __restrict__ b2,
    float* __restrict__ x_out, float* __restrict__ out_col0,
    float* __restrict__ nm, int dnm) {
  constexpr int KB = 5;  // K = DX+129 in (128,160]
  __shared__ __align__(32) _Float16 sHf[4][8][512];
  __shared__ float sDeg[64];

  const int tid = threadIdx.x;
  const int wave = tid >> 5, lane = tid & 31, m = lane & 15;
  const int base = blockIdx.x * 64;
  const int node = base + wave * 16 + m;

  __builtin_prefetch(agg + (long)node * 128, 0, 1);
  if (tid < 64) sDeg[tid] = deg[base + tid];
  __syncthreads();
  float dval = sDeg[wave * 16 + m];
  float invd = 1.f / fmaxf(dval, 1.f);

  // A fragments for full K in registers
  v16h aF[KB];
#pragma unroll
  for (int kb = 0; kb < KB; ++kb) {
#pragma unroll
    for (int i = 0; i < 16; ++i) {
      int k = kb * 32 + frag_k(lane, i);
      float v = 0.f;
      if (k < DX)             v = x[node * DX + k];
      else if (k < DX + 128)  v = agg[(long)node * 128 + (k - DX)] * invd;
      else if (k == DX + 128) v = dval;
      aF[kb][i] = (_Float16)v;
    }
  }

#pragma unroll
  for (int nt = 0; nt < 16; ++nt) {
    v8f acc = {};
#pragma unroll
    for (int kb = 0; kb < KB; ++kb) {
      v16h b = glbB(&wf1[((nt * KB + kb) * 32 + lane) * 16]);
      acc = wmma16(aF[kb], b, acc);
    }
    float bias = b1[nt * 16 + m];
    int n = nt * 16 + m, kb2 = n >> 5, ko = n & 31, h, ii;
    inv_frag(ko, h, ii);
#pragma unroll
    for (int rr = 0; rr < 8; ++rr) {
      int mrow = (lane < 16) ? rr : (8 + rr);
      float v = acc[rr] + bias;
      sHf[wave][kb2][(mrow + 16 * h) * 16 + ii] = (_Float16)(v > 0.f ? v : 0.f);
    }
  }

  // X2 = Hn @ W2 + b (one N tile, cols < DOUT valid)
  v8f a2 = {};
#pragma unroll
  for (int kb = 0; kb < 8; ++kb) {
    v16h a = ldsA(&sHf[wave][kb][lane * 16]);
    v16h b = glbB(&wf2[(kb * 32 + lane) * 16]);
    a2 = wmma16(a, b, a2);
  }
  float bias2 = (m < DOUT) ? b2[m] : 0.f;
#pragma unroll
  for (int rr = 0; rr < 8; ++rr) {
    int mrow = (lane < 16) ? rr : (8 + rr);
    if (m < DOUT) {
      float v = a2[rr] + bias2;
      int nd = base + wave * 16 + mrow;
      if (WRITE_X)  x_out[(long)nd * DOUT + m] = v;
      if (WRITE_NM) atomicAdd(&nm[(nd / NODES_C) * dnm + m], v * (1.f / NODES_C));
      if (WRITE_OUT) { if (m == 0) out_col0[nd] = v; }
    }
  }
}

// ----------------------------- global MLP (tiny) ---------------------------
__launch_bounds__(128)
__global__ void glob_kernel(const float* __restrict__ u, int du,
                            const float* __restrict__ nm, int dnm,
                            const float* __restrict__ w1, const float* __restrict__ b1,
                            const float* __restrict__ w2, const float* __restrict__ b2,
                            int dout, float* __restrict__ u_out) {
  __shared__ float sIn[16];
  __shared__ float sh[128];
  int g = blockIdx.x, t = threadIdx.x;
  int din = du + dnm;
  if (t < din) sIn[t] = (t < du) ? u[g * du + t] : nm[g * dnm + (t - du)];
  __syncthreads();
  float acc = b1[t];
  for (int k = 0; k < din; ++k) acc += sIn[k] * w1[k * 128 + t];
  sh[t] = fmaxf(acc, 0.f);
  __syncthreads();
  if (t < dout) {
    float o = b2[t];
    for (int h = 0; h < 128; ++h) o += sh[h] * w2[h * dout + t];
    u_out[g * dout + t] = o;
  }
}

// --------------------------------- host ------------------------------------

extern "C" void kernel_launch(void* const* d_in, const int* in_sizes, int n_in,
                              void* d_out, int out_size, void* d_ws, size_t ws_size,
                              hipStream_t stream) {
  (void)in_sizes; (void)n_in; (void)out_size; (void)ws_size;
  const float* F = (const float*)d_in[0];
  // params flat order: stacks {p1,p2,v1,v2} x mlps {edge,node1,node2,glob} x {w1,b1,w2,b2}
  auto P = [&](int s, int mlp, int t) -> const float* {
    return (const float*)d_in[1 + s * 16 + mlp * 4 + t];
  };

  char* wsb = (char*)d_ws;
  size_t off = 0;
  auto alloc = [&](size_t bytes) -> void* {
    off = (off + 255) & ~(size_t)255;
    void* p = wsb + off;
    off += bytes;
    return p;
  };

  float* x0  = (float*)alloc((size_t)NN_C * 5 * 4);
  float* x1  = (float*)alloc((size_t)NN_C * 10 * 4);
  float* e0  = (float*)alloc((size_t)NE_C * 1 * 4);
  float* e1  = (float*)alloc((size_t)NE_C * 2 * 4);
  float* u0  = (float*)alloc((size_t)B_C * 6 * 4);
  float* u1  = (float*)alloc((size_t)B_C * 12 * 4);
  int*   row = (int*)alloc((size_t)NE_C * 4);
  int*   col = (int*)alloc((size_t)NE_C * 4);
  float* deg = (float*)alloc((size_t)NN_C * 4);
  float* agg = (float*)alloc((size_t)NN_C * 128 * 4);
  float* nm  = (float*)alloc((size_t)B_C * 10 * 4);

  // per-stack dims (stacks 0,2 = layer1; 1,3 = layer2)
  static const int K1s[2] = {17, 34}, KB1s[2] = {1, 2};
  static const int dEs[2] = {2, 1};
  static const int K2s[2] = {7, 11};
  static const int Kns[2] = {134, 139};
  static const int dOs[2] = {10, 1};

  struct WL { _Float16 *ew1, *ew2, *n1w1, *n1w2, *n2w1, *n2w2; };
  WL W[4];
  auto swz = [&](const float* src, _Float16* dst, int K, int N, int KB, int NT) {
    int total = NT * KB * 512;
    k_swz<<<(total + 255) / 256, 256, 0, stream>>>(src, dst, K, N, KB, NT);
  };
  for (int s = 0; s < 4; ++s) {
    int l = s & 1;
    W[s].ew1  = (_Float16*)alloc((size_t)KB1s[l] * 8 * 512 * 2);
    W[s].ew2  = (_Float16*)alloc((size_t)4 * 512 * 2);
    W[s].n1w1 = (_Float16*)alloc((size_t)8 * 512 * 2);
    W[s].n1w2 = (_Float16*)alloc((size_t)4 * 8 * 512 * 2);
    W[s].n2w1 = (_Float16*)alloc((size_t)5 * 16 * 512 * 2);
    W[s].n2w2 = (_Float16*)alloc((size_t)8 * 512 * 2);
    swz(P(s, 0, 0), W[s].ew1,  K1s[l], 128,    KB1s[l], 8);
    swz(P(s, 0, 2), W[s].ew2,  128,    dEs[l], 4,       1);
    swz(P(s, 1, 0), W[s].n1w1, K2s[l], 128,    1,       8);
    swz(P(s, 1, 2), W[s].n1w2, 128,    128,    4,       8);
    swz(P(s, 2, 0), W[s].n2w1, Kns[l], 256,    5,       16);
    swz(P(s, 2, 2), W[s].n2w2, 256,    dOs[l], 8,       1);
  }

  auto zero = [&](float* p, int n) {
    k_zero<<<(n + 255) / 256, 256, 0, stream>>>(p, n);
  };

  zero(deg, NN_C);
  k_build_x0<<<(NN_C + 255) / 256, 256, 0, stream>>>(F, x0);
  k_build_edges<<<(NE_C + 255) / 256, 256, 0, stream>>>(F, e0, row, col, deg);
  k_build_u0<<<(B_C * 6 + 255) / 256, 256, 0, stream>>>(F, u0);

  float* outF = (float*)d_out;
  for (int st = 0; st < 2; ++st) {  // st=0: policy (p1,p2); st=1: value (v1,v2)
    int sA = st * 2, sB = st * 2 + 1;
    // ---- layer 1 ----
    zero(agg, NN_C * 128);
    zero(nm, B_C * 10);
    edge_node1_kernel<5, 1, 6, true><<<NE_C / 128, 256, 0, stream>>>(
        x0, e0, u0, row, col,
        W[sA].ew1, P(sA, 0, 1), W[sA].ew2, P(sA, 0, 3),
        W[sA].n1w1, P(sA, 1, 1), W[sA].n1w2, P(sA, 1, 3),
        e1, agg);
    node2_kernel<5, 10, true, true, false><<<NN_C / 64, 128, 0, stream>>>(
        x0, agg, deg, W[sA].n2w1, P(sA, 2, 1), W[sA].n2w2, P(sA, 2, 3),
        x1, nullptr, nm, 10);
    glob_kernel<<<B_C, 128, 0, stream>>>(
        u0, 6, nm, 10, P(sA, 3, 0), P(sA, 3, 1), P(sA, 3, 2), P(sA, 3, 3), 12, u1);
    // ---- layer 2 ----
    zero(agg, NN_C * 128);
    edge_node1_kernel<10, 2, 12, false><<<NE_C / 128, 256, 0, stream>>>(
        x1, e1, u1, row, col,
        W[sB].ew1, P(sB, 0, 1), W[sB].ew2, P(sB, 0, 3),
        W[sB].n1w1, P(sB, 1, 1), W[sB].n1w2, P(sB, 1, 3),
        nullptr, agg);
    node2_kernel<10, 1, false, false, true><<<NN_C / 64, 128, 0, stream>>>(
        x1, agg, deg, W[sB].n2w1, P(sB, 2, 1), W[sB].n2w2, P(sB, 2, 3),
        nullptr, outF + (size_t)st * NN_C, nullptr, 0);
  }
}
